// GraphModel_26216480375265
// MI455X (gfx1250) — compile-verified, matrix operand order checked by reading, not run on hardware
//
#include <hip/hip_runtime.h>
#include <hip/hip_bf16.h>
#include <stdint.h>

#define DIM 256
#define HID 128

typedef __attribute__((ext_vector_type(16))) __bf16 v16bf;
typedef __attribute__((ext_vector_type(8)))  float  v8f;

__device__ __forceinline__ unsigned short f32_to_bf16(float f) {
  union { float f; uint32_t u; } v; v.f = f;
  uint32_t r = v.u + 0x7FFFu + ((v.u >> 16) & 1u);   // round-to-nearest-even
  return (unsigned short)(r >> 16);
}

// ---------------- encoder: out[row][d] = relu(in[row]*w1 + b1) @ w2[:,d] + b2[d]
__global__ void enc_kernel(const float* __restrict__ in, const float* __restrict__ w1,
                           const float* __restrict__ b1, const float* __restrict__ w2,
                           const float* __restrict__ b2, float* __restrict__ out) {
  __shared__ float hid[HID];
  const int row = blockIdx.x;
  const int t = threadIdx.x;
  const float xv = in[row];
  if (t < HID) hid[t] = fmaxf(fmaf(xv, w1[t], b1[t]), 0.0f);
  __syncthreads();
  float acc = b2[t];
  #pragma unroll 8
  for (int k = 0; k < HID; ++k) acc = fmaf(hid[k], w2[k * DIM + t], acc);
  out[(size_t)row * DIM + t] = acc;
}

// ---------------- per-edge message + scatter: agg[dst] += relu(h[src]+e) + eps
// one wave32 per edge, 8 floats per lane via two float4 loads
__global__ void msg_kernel(const float* __restrict__ H, const float* __restrict__ E,
                           const int* __restrict__ src, const int* __restrict__ dst,
                           float* __restrict__ agg, int nEdges) {
  const int edge = (int)(((size_t)blockIdx.x * blockDim.x + threadIdx.x) >> 5);
  const int lane = threadIdx.x & 31;
  if (edge >= nEdges) return;
  const int s = src[edge];
  const int d = dst[edge];
  const float4* hv = (const float4*)(H + (size_t)s * DIM);
  const float4* ev = (const float4*)(E + (size_t)edge * DIM);
  float* ap = agg + (size_t)d * DIM;
  #pragma unroll
  for (int i = 0; i < 2; ++i) {
    const int q = lane + i * 32;          // float4 index within the 256-wide row
    const float4 h4 = hv[q];
    const float4 e4 = ev[q];
    atomicAdd(ap + q * 4 + 0, fmaxf(h4.x + e4.x, 0.0f) + 1e-7f);
    atomicAdd(ap + q * 4 + 1, fmaxf(h4.y + e4.y, 0.0f) + 1e-7f);
    atomicAdd(ap + q * 4 + 2, fmaxf(h4.z + e4.z, 0.0f) + 1e-7f);
    atomicAdd(ap + q * 4 + 3, fmaxf(h4.w + e4.w, 0.0f) + 1e-7f);
  }
}

// ---------------- f32 -> bf16 elementwise (4 per thread)
__global__ void cvt_bf16_kernel(const float* __restrict__ in,
                                unsigned short* __restrict__ out, int n4) {
  const int i = blockIdx.x * blockDim.x + threadIdx.x;
  if (i >= n4) return;
  const float4 v = ((const float4*)in)[i];
  union { unsigned short s[4]; uint2 u; } o;
  o.s[0] = f32_to_bf16(v.x);
  o.s[1] = f32_to_bf16(v.y);
  o.s[2] = f32_to_bf16(v.z);
  o.s[3] = f32_to_bf16(v.w);
  ((uint2*)out)[i] = o.u;
}

// ---------------- transpose + convert all layer weights: Wt[l][n][k] = bf16(W[l][k][n])
__global__ void wconv_kernel(const float* __restrict__ W,
                             unsigned short* __restrict__ Wt, int total) {
  const int idx = blockIdx.x * blockDim.x + threadIdx.x;
  if (idx >= total) return;
  const int l = idx >> 16;           // DIM*DIM == 65536 per layer
  const int rem = idx & 0xFFFF;
  const int n = rem >> 8;
  const int k = rem & 0xFF;
  Wt[idx] = f32_to_bf16(W[(size_t)l * (DIM * DIM) + k * DIM + n]);
}

// ---------------- WMMA GEMM: C[M,256] = A[M,256](bf16) x W[256,256] + bias
// A row-major bf16; Bt[n][k] = W[k][n] (bf16). One 16x16 tile per wave, K-steps of 32.
__global__ void gemm_bf16_kernel(const unsigned short* __restrict__ A,
                                 const unsigned short* __restrict__ Bt,
                                 const float* __restrict__ bias,
                                 float* __restrict__ C, int M) {
  const int lane = threadIdx.x & 31;
  const int wave = threadIdx.x >> 5;                 // 0..7
  const int m0 = blockIdx.x * 16;
  const int n0 = (blockIdx.y * 8 + wave) * 16;
  const int l15 = lane & 15;
  const int khalf = (lane >> 4) << 3;                // 0 or 8: ISA 16-bit A-frag layout
  int row = m0 + l15; if (row >= M) row = M - 1;     // clamp; EXEC stays all-1s for WMMA
  const int col = n0 + l15;
  const __bf16* Ab = (const __bf16*)A;
  const __bf16* Bb = (const __bf16*)Bt;
  union Frag { uint4 u[2]; v16bf v; };
  v8f acc = {};
  #pragma unroll
  for (int k0 = 0; k0 < DIM; k0 += 32) {
    Frag fa, fb;
    // lane<16: K[k0..k0+7] + K[k0+16..k0+23]; lane>=16: shifted by 8 (khalf)
    fa.u[0] = *(const uint4*)(Ab + (size_t)row * DIM + k0 + khalf);
    fa.u[1] = *(const uint4*)(Ab + (size_t)row * DIM + k0 + 16 + khalf);
    fb.u[0] = *(const uint4*)(Bb + (size_t)col * DIM + k0 + khalf);
    fb.u[1] = *(const uint4*)(Bb + (size_t)col * DIM + k0 + 16 + khalf);
    acc = __builtin_amdgcn_wmma_f32_16x16x32_bf16(false, fa.v, false, fb.v,
                                                  (short)0, acc, false, false);
  }
  const float bv = bias[col];
  #pragma unroll
  for (int i = 0; i < 8; ++i) {
    const int m = m0 + i + ((lane >> 4) << 3);       // C/D layout: lanes 16-31 hold M+8
    if (m < M) C[(size_t)m * DIM + col] = acc[i] + bv;
  }
}

// ---------------- output head: out[M,3] = H @ out_w + out_b (one wave per node)
__global__ void head_kernel(const float* __restrict__ H, const float* __restrict__ w,
                            const float* __restrict__ b, float* __restrict__ out, int M) {
  const int node = (int)(((size_t)blockIdx.x * blockDim.x + threadIdx.x) >> 5);
  const int lane = threadIdx.x & 31;
  if (node >= M) return;
  const float* hp = H + (size_t)node * DIM;
  float a0 = 0.f, a1 = 0.f, a2 = 0.f;
  #pragma unroll
  for (int k = lane; k < DIM; k += 32) {
    const float hv = hp[k];
    a0 = fmaf(hv, w[k * 3 + 0], a0);
    a1 = fmaf(hv, w[k * 3 + 1], a1);
    a2 = fmaf(hv, w[k * 3 + 2], a2);
  }
  #pragma unroll
  for (int off = 16; off > 0; off >>= 1) {
    a0 += __shfl_down(a0, off, 32);
    a1 += __shfl_down(a1, off, 32);
    a2 += __shfl_down(a2, off, 32);
  }
  if (lane == 0) {
    out[(size_t)node * 3 + 0] = a0 + b[0];
    out[(size_t)node * 3 + 1] = a1 + b[1];
    out[(size_t)node * 3 + 2] = a2 + b[2];
  }
}

extern "C" void kernel_launch(void* const* d_in, const int* in_sizes, int n_in,
                              void* d_out, int out_size, void* d_ws, size_t ws_size,
                              hipStream_t stream) {
  const float* x         = (const float*)d_in[0];
  const float* edge_attr = (const float*)d_in[1];
  const int*   eidx      = (const int*)  d_in[2];
  const float* node_w1   = (const float*)d_in[3];
  const float* node_b1   = (const float*)d_in[4];
  const float* node_w2   = (const float*)d_in[5];
  const float* node_b2   = (const float*)d_in[6];
  const float* edge_w1   = (const float*)d_in[7];
  const float* edge_b1   = (const float*)d_in[8];
  const float* edge_w2   = (const float*)d_in[9];
  const float* edge_b2   = (const float*)d_in[10];
  const float* gnn_w     = (const float*)d_in[11];
  const float* gnn_b     = (const float*)d_in[12];
  const float* out_w     = (const float*)d_in[13];
  const float* out_b     = (const float*)d_in[14];

  const int N = in_sizes[0];                    // 30000 nodes
  const int E = in_sizes[1];                    // 300000 edges
  const int L = in_sizes[11] / (DIM * DIM);     // 4 layers

  // workspace layout (all 256B-aligned for these sizes)
  char* ws = (char*)d_ws;
  size_t off = 0;
  float* Ebuf = (float*)(ws + off);            off += (size_t)E * DIM * sizeof(float);
  float* H    = (float*)(ws + off);            off += (size_t)N * DIM * sizeof(float);
  float* AGG  = (float*)(ws + off);            off += (size_t)N * DIM * sizeof(float);
  unsigned short* Abf = (unsigned short*)(ws + off); off += (size_t)N * DIM * sizeof(unsigned short);
  unsigned short* Wbf = (unsigned short*)(ws + off);

  // encoders (edge encoder computed once, reused by all 4 layers)
  enc_kernel<<<N, DIM, 0, stream>>>(x, node_w1, node_b1, node_w2, node_b2, H);
  enc_kernel<<<E, DIM, 0, stream>>>(edge_attr, edge_w1, edge_b1, edge_w2, edge_b2, Ebuf);

  // weights -> bf16, transposed so GEMM B-fragments are contiguous 16B loads
  const int wtotal = L * DIM * DIM;
  wconv_kernel<<<(wtotal + 255) / 256, 256, 0, stream>>>(gnn_w, Wbf, wtotal);

  const int* src = eidx;        // edge_index[0]
  const int* dst = eidx + E;    // edge_index[1]

  for (int l = 0; l < L; ++l) {
    // agg starts as h (folds the "+ x_dst" of the update into the aggregation)
    hipMemcpyAsync(AGG, H, (size_t)N * DIM * sizeof(float),
                   hipMemcpyDeviceToDevice, stream);
    msg_kernel<<<(E + 7) / 8, 256, 0, stream>>>(H, Ebuf, src, dst, AGG, E);
    const int n4 = (N * DIM) / 4;
    cvt_bf16_kernel<<<(n4 + 255) / 256, 256, 0, stream>>>(AGG, Abf, n4);
    dim3 g((N + 15) / 16, (DIM / 16) / 8);   // 1875 x 2, 8 waves/block -> one 16x16 tile each
    gemm_bf16_kernel<<<g, 256, 0, stream>>>(Abf, Wbf + (size_t)l * DIM * DIM,
                                            gnn_b + l * DIM, H, N);
  }

  head_kernel<<<((size_t)N * 32 + 255) / 256, 256, 0, stream>>>(H, out_w, out_b,
                                                                (float*)d_out, N);
}
